// Layer3bitFaster_83356725281468
// MI455X (gfx1250) — matrix operand order repaired and checked by reading, not compiled, not run on hardware
//
#include <hip/hip_runtime.h>

#define MM 64
#define KK 4096
#define NN 11008
#define ZERO_PT 4
#define KSPLIT 8
#define JSTEPS (KK / 32)          // 128 k-steps of 32
#define JCHUNK (JSTEPS / KSPLIT)  // 16 k-steps per wave
#define NTILES (NN / 16)          // 688 column tiles
#define WPB 8                     // waves per block
// LDS: 4 M-tiles x JCHUNK k-steps x (32 lanes x 16 bf16) = 64 KB
#define LDS_BYTES (4 * JCHUNK * 32 * 16 * 2)

typedef __attribute__((ext_vector_type(16))) __bf16 v16bf;
typedef __attribute__((ext_vector_type(8)))  float  v8f;

__global__ void zero_out_kernel(float* __restrict__ out, int n) {
    int i = blockIdx.x * blockDim.x + threadIdx.x;
    if (i < n) out[i] = 0.0f;
}

// Repack A (f32, 64x4096 row-major) into bf16 WMMA A-fragments in d_ws.
// Fragment layout used here, contiguous k order (16-bit A, 16x32):
//   lanes 0-15: M = lane,     elems 0-15 -> K=0..15
//   lanes 16-31: M = lane-16, elems 0-15 -> K=16..31
// ws index = ((t*128 + j)*32 + lane)*16 + e
__global__ void prep_A_kernel(const float* __restrict__ A, __bf16* __restrict__ wsA) {
    int idx = blockIdx.x * blockDim.x + threadIdx.x;   // 0 .. 4*128*32*16-1
    int e    = idx & 15;
    int lane = (idx >> 4) & 31;
    int j    = (idx >> 9) & 127;
    int t    = idx >> 16;
    int half = lane >> 4;
    int m = t * 16 + (lane & 15);
    int k = j * 32 + e + 16 * half;
    wsA[idx] = (__bf16)A[m * KK + k];
}

__global__ void __launch_bounds__(256)
gemm3bit_kernel(const int* __restrict__ q, const float* __restrict__ s,
                const __bf16* __restrict__ wsA, float* __restrict__ out) {
    extern __shared__ char smem[];
    const int tid  = threadIdx.x;
    const int lane = tid & 31;
    const int wave = tid >> 5;
    const int half = lane >> 4;        // 0 or 1
    const int nIdx = lane & 15;
    const int n = (blockIdx.x * WPB + wave) * 16 + nIdx;   // < 11008 always
    const int jbeg = blockIdx.y * JCHUNK;

    // ---- async DMA stage of this K-chunk's A fragments into LDS ----
    // LDS fragment (t, jj, lane) = v16bf at index (t*JCHUNK + jj)*32 + lane.
    // GLOBAL_LOAD_ASYNC_TO_LDS_B128: VDST = LDS byte offset, VADDR = 64b addr,
    // tracked by ASYNCcnt (no VGPR data movement).
    {
        const uint4* __restrict__ src = (const uint4*)wsA;  // 64 uint4 per (t,j)
        const int PER_T = JCHUNK * 64;                      // uint4 per M-tile segment
#pragma unroll
        for (int i = tid; i < LDS_BYTES / 16; i += 256) {
            int t = i / PER_T;
            int r = i - t * PER_T;
            const uint4* gsrc = src + (size_t)(t * JSTEPS + jbeg) * 64 + r;
            unsigned lds_off = (unsigned)i * 16u;
            asm volatile("global_load_async_to_lds_b128 %0, %1, off"
                         :: "v"(lds_off), "v"(gsrc) : "memory");
        }
#if __has_builtin(__builtin_amdgcn_s_wait_asynccnt)
        __builtin_amdgcn_s_wait_asynccnt(0);
#else
        asm volatile("s_wait_asynccnt 0x0" ::: "memory");
#endif
    }
    __syncthreads();
    const v16bf* __restrict__ lA = (const v16bf*)smem;

    v8f acc0 = {}, acc1 = {}, acc2 = {}, acc3 = {};

    // q stream pointer: advances 32 rows (32*NN ints) per k-step; element e of
    // the B fragment is at row K = e + 16*half (24-bit immediate offsets).
    const int* qptr = q + (size_t)(jbeg * 32) * NN + n;

    // ---- software pipeline: preload k-step jbeg ----
    int qv[16];
#pragma unroll
    for (int e = 0; e < 16; ++e)
        qv[e] = __builtin_nontemporal_load(qptr + (size_t)(e + 16 * half) * NN);
    float scale = s[(size_t)(jbeg >> 1) * NN + n];

    for (int jj = 0; jj < JCHUNK; ++jj) {
        // dequantize the codes loaded last iteration -> B fragment (32x16 K x N)
        v16bf b;
#pragma unroll
        for (int e = 0; e < 16; ++e)
            b[e] = (__bf16)((float)(qv[e] - ZERO_PT) * scale);

        // issue next k-step's q loads before the WMMAs (clamped on last iter)
        const bool more = (jj + 1) < JCHUNK;
        const int*   qnext = more ? (qptr + (size_t)32 * NN) : (q + n);
        const float* snext = more ? (s + (size_t)((jbeg + jj + 1) >> 1) * NN + n)
                                  : (s + n);
#pragma unroll
        for (int e = 0; e < 16; ++e)
            qv[e] = __builtin_nontemporal_load(qnext + (size_t)(e + 16 * half) * NN);
        scale = *snext;
        qptr += (size_t)32 * NN;

        // A fragments from LDS (low latency, no global traffic)
        v16bf a0 = lA[(0 * JCHUNK + jj) * 32 + lane];
        v16bf a1 = lA[(1 * JCHUNK + jj) * 32 + lane];
        v16bf a2 = lA[(2 * JCHUNK + jj) * 32 + lane];
        v16bf a3 = lA[(3 * JCHUNK + jj) * 32 + lane];

        acc0 = __builtin_amdgcn_wmma_f32_16x16x32_bf16(false, a0, false, b,
                                                       (short)0, acc0, false, false);
        acc1 = __builtin_amdgcn_wmma_f32_16x16x32_bf16(false, a1, false, b,
                                                       (short)0, acc1, false, false);
        acc2 = __builtin_amdgcn_wmma_f32_16x16x32_bf16(false, a2, false, b,
                                                       (short)0, acc2, false, false);
        acc3 = __builtin_amdgcn_wmma_f32_16x16x32_bf16(false, a3, false, b,
                                                       (short)0, acc3, false, false);
    }

    // C/D layout: VGPR r -> M = r (lanes 0-15) / r+8 (lanes 16-31), N = lane&15
#pragma unroll
    for (int r = 0; r < 8; ++r) {
        const int mbase = r + 8 * half;
        atomicAdd(&out[(size_t)(mbase +  0) * NN + n], acc0[r]);
        atomicAdd(&out[(size_t)(mbase + 16) * NN + n], acc1[r]);
        atomicAdd(&out[(size_t)(mbase + 32) * NN + n], acc2[r]);
        atomicAdd(&out[(size_t)(mbase + 48) * NN + n], acc3[r]);
    }
}

extern "C" void kernel_launch(void* const* d_in, const int* in_sizes, int n_in,
                              void* d_out, int out_size, void* d_ws, size_t ws_size,
                              hipStream_t stream) {
    const float* A = (const float*)d_in[0];   // (64, 4096) f32
    const int*   q = (const int*)d_in[1];     // (4096, 11008) int32 codes
    const float* s = (const float*)d_in[2];   // (64, 11008) f32 scales
    float* out = (float*)d_out;               // (64, 11008) f32
    __bf16* wsA = (__bf16*)d_ws;              // 512 KB A fragments

    zero_out_kernel<<<(MM * NN + 255) / 256, 256, 0, stream>>>(out, MM * NN);
    prep_A_kernel<<<(4 * JSTEPS * 32 * 16 + 255) / 256, 256, 0, stream>>>(A, wsA);

    dim3 grid(NTILES / WPB, KSPLIT);          // (86, 8)
    gemm3bit_kernel<<<grid, 256, LDS_BYTES, stream>>>(q, s, wsA, out);
}
